// BinaryConv2dBBCU_44281112822023
// MI455X (gfx1250) — 8-bit matrix operand order repaired and checked by reading, compiled, not run on hardware
//
#include <hip/hip_runtime.h>

typedef int v8i __attribute__((ext_vector_type(8)));

#define C_CH   64
#define H_DIM  256
#define W_DIM  256
#define B_DIM  16
#define HW     (H_DIM * W_DIM)
#define NPIX   ((size_t)B_DIM * HW)
#define KPOS   9
#define WBYTES (KPOS * C_CH * C_CH)

__global__ void pack_weights_kernel(const float* __restrict__ w,
                                    char* __restrict__ wp,
                                    float* __restrict__ scale) {
    int cout = threadIdx.x;
    float acc = 0.0f;
    for (int cin = 0; cin < C_CH; ++cin) {
        for (int k = 0; k < KPOS; ++k) {
            float v = w[((size_t)cout * C_CH + cin) * KPOS + k];
            acc += fabsf(v);
            char sg = (v > 0.0f) ? (char)1 : ((v < 0.0f) ? (char)-1 : (char)0);
            wp[((size_t)k * C_CH + cout) * C_CH + cin] = sg;
        }
    }
    scale[cout] = acc / (float)(C_CH * KPOS);
}

__global__ __launch_bounds__(256) void binarize_pack_kernel(
        const float* __restrict__ x, const float* __restrict__ bias,
        char* __restrict__ s) {
    size_t p = (size_t)blockIdx.x * blockDim.x + threadIdx.x;
    if (p >= NPIX) return;
    size_t n  = p >> 16;
    size_t hw = p & (HW - 1);
    const float* xp = x + n * (size_t)C_CH * HW + hw;
    int packed[16];
#pragma unroll
    for (int g = 0; g < 16; ++g) {
        int v = 0;
#pragma unroll
        for (int j = 0; j < 4; ++j) {
            int c = g * 4 + j;
            float t = xp[(size_t)c * HW] + bias[c];
            int sg = (t > 0.0f) ? 1 : ((t < 0.0f) ? -1 : 0);
            v |= (sg & 0xFF) << (8 * j);
        }
        packed[g] = v;
    }
    int4* dst = (int4*)(s + p * C_CH);
#pragma unroll
    for (int q = 0; q < 4; ++q) {
        int4 t; t.x = packed[q*4+0]; t.y = packed[q*4+1];
        t.z = packed[q*4+2]; t.w = packed[q*4+3];
        dst[q] = t;
    }
}

__global__ __launch_bounds__(256) void bconv_wmma_kernel(
        const char*  __restrict__ s,
        const char*  __restrict__ wp,
        const float* __restrict__ scale,
        const float* __restrict__ x,
        const float* __restrict__ prelu,
        const float* __restrict__ b0,
        const float* __restrict__ b1,
        float* __restrict__ out) {
    __shared__ __align__(16) char  lds_w[WBYTES];
    __shared__ float lds_par[4 * C_CH];
    __shared__ __align__(16) float lds_t[8 * C_CH * 16];

    int tid = threadIdx.x;

#if __has_builtin(__builtin_amdgcn_global_load_async_to_lds_b128)
    {
        typedef int v4i_vs __attribute__((vector_size(16)));
        typedef __attribute__((address_space(1))) v4i_vs* gvp_t;
        typedef __attribute__((address_space(3))) v4i_vs* lvp_t;
        for (int i = tid * 16; i < WBYTES; i += 256 * 16) {
            __builtin_amdgcn_global_load_async_to_lds_b128(
                (gvp_t)(void*)(wp + i), (lvp_t)(void*)(lds_w + i), 0, 0);
        }
        asm volatile("s_wait_asynccnt 0x0" ::: "memory");
    }
#else
    {
        const int4* src = (const int4*)wp;
        int4* dst = (int4*)lds_w;
        for (int i = tid; i < WBYTES / 16; i += 256) dst[i] = src[i];
    }
#endif
    if (tid < C_CH) {
        lds_par[tid]            = scale[tid];
        lds_par[C_CH + tid]     = prelu[tid];
        lds_par[2 * C_CH + tid] = b0[tid];
        lds_par[3 * C_CH + tid] = b1[tid];
    }
    __syncthreads();

    int wave    = tid >> 5;
    int lane    = tid & 31;
    int m       = lane & 15;
    int halfsel = lane >> 4;

    size_t tile    = (size_t)blockIdx.x * 8 + wave;
    size_t pixBase = tile * 16;

    size_t p  = pixBase + (size_t)m;
    size_t n  = p >> 16;
    int    hw = (int)(p & (HW - 1));
    int    h  = hw >> 8;
    int    w  = hw & (W_DIM - 1);

    v8i acc[4];
#pragma unroll
    for (int t = 0; t < 4; ++t)
#pragma unroll
        for (int r = 0; r < 8; ++r) acc[t][r] = 0;

    int krow = m + halfsel * 16;
    const size_t nBase = n * (size_t)HW;

    for (int kh = 0; kh < 3; ++kh) {
        for (int kw = 0; kw < 3; ++kw) {
            int kpos = kh * 3 + kw;
            int hh = h + kh - 1, ww = w + kw - 1;
            bool inb = (hh >= 0) & (hh < H_DIM) & (ww >= 0) & (ww < W_DIM);
            const char* arowb =
                s + (nBase + (size_t)hh * W_DIM + ww) * C_CH + halfsel * 8;
            uint2 z2; z2.x = 0u; z2.y = 0u;
            uint2 c0 = inb ? *(const uint2*)(arowb +  0) : z2;
            uint2 c1 = inb ? *(const uint2*)(arowb + 16) : z2;
            uint2 c2 = inb ? *(const uint2*)(arowb + 32) : z2;
            uint2 c3 = inb ? *(const uint2*)(arowb + 48) : z2;
            v8i a;
            a[0] = (int)c0.x; a[1] = (int)c0.y;
            a[2] = (int)c1.x; a[3] = (int)c1.y;
            a[4] = (int)c2.x; a[5] = (int)c2.y;
            a[6] = (int)c3.x; a[7] = (int)c3.y;
#pragma unroll
            for (int t = 0; t < 4; ++t) {
                const char* bq =
                    lds_w + ((size_t)(kpos * C_CH + t * 16 + m) * C_CH) + halfsel * 8;
                uint2 q0 = *(const uint2*)(bq +  0);
                uint2 q1 = *(const uint2*)(bq + 16);
                uint2 q2 = *(const uint2*)(bq + 32);
                uint2 q3 = *(const uint2*)(bq + 48);
                v8i b;
                b[0] = (int)q0.x; b[1] = (int)q0.y; b[2] = (int)q1.x; b[3] = (int)q1.y;
                b[4] = (int)q2.x; b[5] = (int)q2.y; b[6] = (int)q3.x; b[7] = (int)q3.y;
                acc[t] = __builtin_amdgcn_wmma_i32_16x16x64_iu8(
                    true, a, true, b, acc[t], false, false);
            }
        }
    }

    float* tbuf = lds_t + wave * (C_CH * 16);
#pragma unroll
    for (int t = 0; t < 4; ++t) {
        int   cout = t * 16 + m;
        float sc  = lds_par[cout];
        float ap  = lds_par[C_CH + cout];
        float bb0 = lds_par[2 * C_CH + cout];
        float bb1 = lds_par[3 * C_CH + cout];
#pragma unroll
        for (int r = 0; r < 8; ++r) {
            float y = sc * (float)acc[t][r];
            y += bb0;
            y = (y > 0.0f) ? y : ap * y;
            y += bb1;
            tbuf[cout * 16 + (r + 8 * halfsel)] = y;
        }
    }
    int hw0 = (int)(pixBase & (HW - 1));
    int quad = lane & 3;
#pragma unroll
    for (int co = 0; co < 8; ++co) {
        int cout = co * 8 + (lane >> 2);
        float4 yv = *(float4*)&tbuf[cout * 16 + quad * 4];
        size_t gidx = (n * C_CH + (size_t)cout) * (size_t)HW + hw0 + quad * 4;
        float4 xv = *(const float4*)&x[gidx];
        yv.x += xv.x; yv.y += xv.y; yv.z += xv.z; yv.w += xv.w;
        *(float4*)&out[gidx] = yv;
    }
}

extern "C" void kernel_launch(void* const* d_in, const int* in_sizes, int n_in,
                              void* d_out, int out_size, void* d_ws, size_t ws_size,
                              hipStream_t stream) {
    const float* x      = (const float*)d_in[0];
    const float* mbias  = (const float*)d_in[1];
    const float* wgt    = (const float*)d_in[2];
    const float* prelu  = (const float*)d_in[3];
    const float* pb0    = (const float*)d_in[4];
    const float* pb1    = (const float*)d_in[5];
    float* out = (float*)d_out;

    char*  s_act  = (char*)d_ws;
    char*  wp     = s_act + NPIX * C_CH;
    float* wscale = (float*)(wp + WBYTES);

    pack_weights_kernel<<<1, 64, 0, stream>>>(wgt, wp, wscale);

    binarize_pack_kernel<<<(int)(NPIX / 256), 256, 0, stream>>>(x, mbias, s_act);

    bconv_wmma_kernel<<<(int)(NPIX / 16 / 8), 256, 0, stream>>>(
        s_act, wp, wscale, x, prelu, pb0, pb1, out);
}
